// MPNNet_4887672783316
// MI455X (gfx1250) — matrix operand order repaired and checked by reading, not compiled, not run on hardware
//
#include <hip/hip_runtime.h>
#include <hip/hip_bf16.h>

// ---------------------------------------------------------------------------
// MPNN forward for MI455X (gfx1250, wave32, WMMA).
// Dominant cost: W_edges (40000 x 64 x 64) stored f16 (328 MB), written once,
// read 3x by the message-passing kernel -> ~1.3 GB total ~ 56us @ 23.3 TB/s.
// All dense GEMMs use v_wmma_f32_16x16x32_f16 (fp32 accumulate).
// Workspace use ~392 MB (assumes ws_size is sufficient).
// ---------------------------------------------------------------------------

typedef __attribute__((ext_vector_type(16))) _Float16 v16h;
typedef __attribute__((ext_vector_type(8)))  _Float16 v8h;
typedef __attribute__((ext_vector_type(2)))  _Float16 v2h;
typedef __attribute__((ext_vector_type(8)))  float    v8f;

#define N_NODES  20000
#define N_EDGES  40000
#define N_GRAPHS 800
#define ATOM     64
#define CONV_DIM 128
#define EDGE_DIM 16
#define MLP1     512
#define MLP2     256
#define OUTD     34

__device__ __forceinline__ float sigf(float x) { return 1.0f / (1.0f + __expf(-x)); }

__device__ __forceinline__ unsigned f2ord(float f) {
  unsigned u = __float_as_uint(f);
  return (u & 0x80000000u) ? ~u : (u | 0x80000000u);
}
__device__ __forceinline__ float ord2f(unsigned u) {
  unsigned v = (u & 0x80000000u) ? (u & 0x7FFFFFFFu) : ~u;
  return __uint_as_float(v);
}

// ---------------------------------------------------------------------------
// Generic WMMA GEMM: C[M,N] = act(A[M,K] f16 row-major x BT[N,K] f16 + bias)
// One wave per 16x16 output tile; K stepped by 32 (all our dims are exact).
// A lane layout (ISA 7.12.2, 16-bit A 16x32): lane<16 -> K 0..7,16..23 of
// row M=lane; lane>=16 -> K 8..15,24..31. B (from transposed [N,K] weights):
// same K layout as A for column N=lane&15.
// ---------------------------------------------------------------------------
template <bool RELU, bool WF32, bool WF16>
__global__ __launch_bounds__(256) void k_wmma_gemm(
    const _Float16* __restrict__ A, const _Float16* __restrict__ BT,
    const float* __restrict__ bias, float* __restrict__ C,
    _Float16* __restrict__ C16, int M, int N, int K) {
  const int wave = (blockIdx.x * blockDim.x + threadIdx.x) >> 5;
  const int lane = threadIdx.x & 31;
  const int ntiles = N >> 4;
  const int tiles = (M >> 4) * ntiles;
  if (wave >= tiles) return;
  const int tm = wave / ntiles;
  const int tn = wave - tm * ntiles;

  const int m  = tm * 16 + (lane & 15);
  const int n  = tn * 16 + (lane & 15);
  const int ka = (lane >> 4) * 8;    // A K-half offset within each 16
  const int kb = (lane >> 4) * 8;    // B K-half offset within each 16

  const _Float16* ap = A + (size_t)m * K;
  const _Float16* bp = BT + (size_t)n * K;

  v8f acc = {};
  for (int k0 = 0; k0 < K; k0 += 32) {
    __builtin_prefetch(ap + k0 + 32, 0, 0);   // global_prefetch_b8
    __builtin_prefetch(bp + k0 + 32, 0, 0);
    v8h a0 = *(const v8h*)(ap + k0 + ka);
    v8h a1 = *(const v8h*)(ap + k0 + 16 + ka);
    v16h av, bv;
    v8h b0 = *(const v8h*)(bp + k0 + kb);
    v8h b1 = *(const v8h*)(bp + k0 + 16 + kb);
#pragma unroll
    for (int i = 0; i < 8; ++i) { av[i] = a0[i]; av[i + 8] = a1[i]; }
#pragma unroll
    for (int i = 0; i < 8; ++i) { bv[i] = b0[i]; bv[i + 8] = b1[i]; }
    acc = __builtin_amdgcn_wmma_f32_16x16x32_f16(
        /*neg_a=*/false, av, /*neg_b=*/false, bv,
        /*c_mod=*/(short)0, acc, /*reuse_a=*/false, /*reuse_b=*/false);
  }

  // D layout: VGPR v, lanes 0-15 -> row v, lanes 16-31 -> row v+8; col = lane&15
  const int row0 = tm * 16 + ((lane >> 4) ? 8 : 0);
  const int col  = tn * 16 + (lane & 15);
  const float bval = bias ? bias[col] : 0.0f;
#pragma unroll
  for (int v = 0; v < 8; ++v) {
    float val = acc[v] + bval;
    if (RELU) val = fmaxf(val, 0.0f);
    const size_t idx = (size_t)(row0 + v) * N + col;
    if (WF32) C[idx]   = val;
    if (WF16) C16[idx] = (_Float16)val;
  }
}

// ---------------------------------------------------------------------------
// Small utility / fused kernels
// ---------------------------------------------------------------------------
__global__ void k_zero(float* p, int n) {
  int i = blockIdx.x * blockDim.x + threadIdx.x;
  if (i < n) p[i] = 0.0f;
}

__global__ void k_f32_to_f16(const float* __restrict__ s, _Float16* __restrict__ d, int n) {
  int i = blockIdx.x * blockDim.x + threadIdx.x;
  if (i < n) d[i] = (_Float16)s[i];
}

// dst[C,R] = f16(src[R,C])
__global__ void k_transpose_f16(const float* __restrict__ s, _Float16* __restrict__ d,
                                int R, int C) {
  int i = blockIdx.x * blockDim.x + threadIdx.x;
  if (i >= R * C) return;
  int r = i / C, c = i - r * C;
  d[(size_t)c * R + r] = (_Float16)s[i];
}

// Concatenate LSTM weights along K: Wcat[256,192] = [Wih[256,128] | Whh[256,64]]
__global__ void k_wcat(const float* __restrict__ Wih, const float* __restrict__ Whh,
                       const float* __restrict__ bih, const float* __restrict__ bhh,
                       _Float16* __restrict__ Wcat, float* __restrict__ bcat) {
  int i = blockIdx.x * blockDim.x + threadIdx.x;
  if (i >= 256 * 192) return;
  int n = i / 192, k = i - n * 192;
  Wcat[i] = (_Float16)((k < 128) ? Wih[n * 128 + k] : Whh[n * 64 + (k - 128)]);
  if (i < 256) bcat[i] = bih[i] + bhh[i];
}

// edge MLP1: H[40000,128] = f16(relu(edge_attr @ W_e1 + b_e1)), K=16 (VALU)
__global__ void k_edge1(const float* __restrict__ ea, const float* __restrict__ W,
                        const float* __restrict__ b, _Float16* __restrict__ H) {
  int i = blockIdx.x * blockDim.x + threadIdx.x;
  if (i >= N_EDGES * CONV_DIM) return;
  int e = i >> 7, c = i & 127;
  const float* a = ea + (size_t)e * EDGE_DIM;
  float acc = b[c];
#pragma unroll
  for (int k = 0; k < EDGE_DIM; ++k) acc += a[k] * W[k * CONV_DIM + c];
  H[i] = (_Float16)fmaxf(acc, 0.0f);
}

__global__ void k_deg(const int* __restrict__ dst, float* deg, int E) {
  int e = blockIdx.x * blockDim.x + threadIdx.x;
  if (e < E) atomicAdd(&deg[dst[e]], 1.0f);
}

// Message passing: one wave per edge. msg = x[src] (1x64) @ W_e (64x64 f16).
// Lane owns columns 2*lane, 2*lane+1 -> 128B coalesced row reads. Memory-bound
// (328 MB of W_edges per round), so VALU FMA is the right tool here.
__global__ __launch_bounds__(256) void k_message(
    const float* __restrict__ x, const _Float16* __restrict__ W,
    const int* __restrict__ srcI, const int* __restrict__ dstI,
    float* __restrict__ agg, int E) {
  int wave = (blockIdx.x * blockDim.x + threadIdx.x) >> 5;
  int lane = threadIdx.x & 31;
  if (wave >= E) return;
  int s = srcI[wave], d = dstI[wave];
  float xlo = x[(size_t)s * 64 + lane];
  float xhi = x[(size_t)s * 64 + 32 + lane];
  const _Float16* We = W + (size_t)wave * (ATOM * ATOM);
  float a0 = 0.0f, a1 = 0.0f;
#pragma unroll 4
  for (int dd = 0; dd < 32; ++dd) {
    float xv = __shfl(xlo, dd, 32);
    v2h w = *(const v2h*)(We + dd * 64 + 2 * lane);
    a0 += xv * (float)w[0];
    a1 += xv * (float)w[1];
  }
#pragma unroll 4
  for (int dd = 0; dd < 32; ++dd) {
    float xv = __shfl(xhi, dd, 32);
    v2h w = *(const v2h*)(We + (32 + dd) * 64 + 2 * lane);
    a0 += xv * (float)w[0];
    a1 += xv * (float)w[1];
  }
  atomicAdd(&agg[(size_t)d * 64 + 2 * lane], a0);
  atomicAdd(&agg[(size_t)d * 64 + 2 * lane + 1], a1);
}

// m = relu(agg/deg + b_conv) -> f16 (GEMM A operand)
__global__ void k_mprep(const float* __restrict__ agg, const float* __restrict__ deg,
                        const float* __restrict__ bc, _Float16* __restrict__ m16) {
  int i = blockIdx.x * blockDim.x + threadIdx.x;
  if (i >= N_NODES * ATOM) return;
  int node = i >> 6, c = i & 63;
  float dv = fmaxf(deg[node], 1.0f);
  m16[i] = (_Float16)fmaxf(agg[i] / dv + bc[c], 0.0f);
}

// GRU gate fusion (gates [r,z,n])
__global__ void k_gru(const float* __restrict__ gi, const float* __restrict__ gh,
                      float* __restrict__ h32, _Float16* __restrict__ h16) {
  int i = blockIdx.x * blockDim.x + threadIdx.x;
  if (i >= N_NODES * ATOM) return;
  int node = i >> 6, c = i & 63;
  const float* gin = gi + (size_t)node * 192;
  const float* ghn = gh + (size_t)node * 192;
  float r = sigf(gin[c] + ghn[c]);
  float z = sigf(gin[64 + c] + ghn[64 + c]);
  float n = tanhf(gin[128 + c] + r * ghn[128 + c]);
  float h = h32[i];
  float hn = (1.0f - z) * n + z * h;
  h32[i] = hn;
  h16[i] = (_Float16)hn;
}

// Xcat[800,192] = f16([q_star | hl]) for the fused Set2Set LSTM GEMM
__global__ void k_xcat(const float* __restrict__ qs, const float* __restrict__ hl,
                       _Float16* __restrict__ xc) {
  int i = blockIdx.x * blockDim.x + threadIdx.x;
  if (i >= N_GRAPHS * 192) return;
  int b = i / 192, c = i - b * 192;
  float v = (c < 128) ? qs[b * 128 + c] : hl[b * 64 + (c - 128)];
  xc[i] = (_Float16)v;
}

// LSTM gate fusion (gates [i,f,g,o]); writes q into q_star[:, :64] and zeroes
// q_star[:, 64:128] ahead of the attention readout accumulation.
__global__ void k_lstm(const float* __restrict__ g, float* __restrict__ cl,
                       float* __restrict__ hl, float* __restrict__ qs) {
  int i = blockIdx.x * blockDim.x + threadIdx.x;
  if (i >= N_GRAPHS * ATOM) return;
  int b = i >> 6, c = i & 63;
  const float* gb = g + (size_t)b * 256;
  float ig = sigf(gb[c]);
  float fg = sigf(gb[64 + c]);
  float cg = tanhf(gb[128 + c]);
  float og = sigf(gb[192 + c]);
  float cn = fg * cl[i] + ig * cg;
  cl[i] = cn;
  float hn = og * tanhf(cn);
  hl[i] = hn;
  qs[b * 128 + c] = hn;        // q
  qs[b * 128 + 64 + c] = 0.0f; // clear r_read accumulator slot
}

// e[node] = dot(x[node], hl[batch[node]]); segment max via ordered-uint atomicMax
__global__ __launch_bounds__(256) void k_dot(
    const float* __restrict__ x, const float* __restrict__ hl,
    const int* __restrict__ batch, float* __restrict__ e, unsigned* __restrict__ emax) {
  int wave = (blockIdx.x * blockDim.x + threadIdx.x) >> 5;
  int lane = threadIdx.x & 31;
  if (wave >= N_NODES) return;
  int b = batch[wave];
  float p = x[(size_t)wave * 64 + lane] * hl[(size_t)b * 64 + lane] +
            x[(size_t)wave * 64 + 32 + lane] * hl[(size_t)b * 64 + 32 + lane];
  for (int o = 16; o; o >>= 1) p += __shfl_xor(p, o, 32);
  if (lane == 0) {
    e[wave] = p;
    atomicMax(&emax[b], f2ord(p));
  }
}

__global__ void k_expsum(float* __restrict__ e, const unsigned* __restrict__ emax,
                         const int* __restrict__ batch, float* __restrict__ zden) {
  int i = blockIdx.x * blockDim.x + threadIdx.x;
  if (i >= N_NODES) return;
  int b = batch[i];
  float v = __expf(e[i] - ord2f(emax[b]));
  e[i] = v;
  atomicAdd(&zden[b], v);
}

// r_read accumulated directly into q_star[:, 64:128]
__global__ __launch_bounds__(256) void k_attn(
    const float* __restrict__ ee, const float* __restrict__ zden,
    const int* __restrict__ batch, const float* __restrict__ x, float* __restrict__ qs) {
  int wave = (blockIdx.x * blockDim.x + threadIdx.x) >> 5;
  int lane = threadIdx.x & 31;
  if (wave >= N_NODES) return;
  int b = batch[wave];
  float a = ee[wave] / zden[b];
  atomicAdd(&qs[(size_t)b * 128 + 64 + lane], a * x[(size_t)wave * 64 + lane]);
  atomicAdd(&qs[(size_t)b * 128 + 96 + lane], a * x[(size_t)wave * 64 + 32 + lane]);
}

// Training-mode BatchNorm over 800 rows x 128 cols; one block, thread per col.
__global__ void k_bn(const float* __restrict__ qs, const float* __restrict__ gamma,
                     const float* __restrict__ beta, _Float16* __restrict__ y16) {
  int c = threadIdx.x;
  if (c >= 128) return;
  float s = 0.0f, s2 = 0.0f;
  for (int b = 0; b < N_GRAPHS; ++b) {
    float v = qs[b * 128 + c];
    s += v; s2 += v * v;
  }
  float mu = s * (1.0f / N_GRAPHS);
  float var = s2 * (1.0f / N_GRAPHS) - mu * mu;
  float inv = rsqrtf(var + 1e-5f) * gamma[c];
  float be = beta[c];
  for (int b = 0; b < N_GRAPHS; ++b)
    y16[b * 128 + c] = (_Float16)((qs[b * 128 + c] - mu) * inv + be);
}

// Final head: [800,256] @ [256,34] + b (N=34 not WMMA-friendly -> VALU)
__global__ void k_head(const float* __restrict__ y2, const float* __restrict__ Wp,
                       const float* __restrict__ bp, float* __restrict__ out) {
  int i = blockIdx.x * blockDim.x + threadIdx.x;
  if (i >= N_GRAPHS * OUTD) return;
  int b = i / OUTD, o = i - b * OUTD;
  float acc = bp[o];
  const float* yb = y2 + (size_t)b * MLP2;
  for (int k = 0; k < MLP2; ++k) acc += yb[k] * Wp[k * OUTD + o];
  out[i] = acc;
}

// ---------------------------------------------------------------------------
static inline int cdiv(long long a, long long b) { return (int)((a + b - 1) / b); }

extern "C" void kernel_launch(void* const* d_in, const int* in_sizes, int n_in,
                              void* d_out, int out_size, void* d_ws, size_t ws_size,
                              hipStream_t stream) {
  (void)in_sizes; (void)n_in; (void)out_size; (void)ws_size;
  const float* x_feat   = (const float*)d_in[0];
  const float* edge_attr= (const float*)d_in[1];
  const int*   edge_idx = (const int*)  d_in[2];
  const int*   batch    = (const int*)  d_in[3];
  const float* W_lin = (const float*)d_in[4];  const float* b_lin = (const float*)d_in[5];
  const float* W_e1  = (const float*)d_in[6];  const float* b_e1  = (const float*)d_in[7];
  const float* W_e2  = (const float*)d_in[8];  const float* b_e2  = (const float*)d_in[9];
  const float* b_conv= (const float*)d_in[10];
  const float* gWih  = (const float*)d_in[11]; const float* gWhh  = (const float*)d_in[12];
  const float* gbih  = (const float*)d_in[13]; const float* gbhh  = (const float*)d_in[14];
  const float* lWih  = (const float*)d_in[15]; const float* lWhh  = (const float*)d_in[16];
  const float* lbih  = (const float*)d_in[17]; const float* lbhh  = (const float*)d_in[18];
  const float* bn_g  = (const float*)d_in[19]; const float* bn_b  = (const float*)d_in[20];
  const float* W_m1  = (const float*)d_in[21]; const float* b_m1  = (const float*)d_in[22];
  const float* W_m2  = (const float*)d_in[23]; const float* b_m2  = (const float*)d_in[24];
  const float* W_p   = (const float*)d_in[25]; const float* b_p   = (const float*)d_in[26];
  float* out = (float*)d_out;

  const int* srcI = edge_idx;
  const int* dstI = edge_idx + N_EDGES;

  // Workspace bump allocator (256B aligned)
  char* ws = (char*)d_ws; size_t off = 0;
  auto alloc = [&](size_t bytes) -> void* {
    off = (off + 255) & ~(size_t)255;
    void* p = ws + off; off += bytes; return p;
  };
  _Float16* xfeat16 = (_Float16*)alloc((size_t)N_NODES * ATOM * 2);
  float*    h32     = (float*)   alloc((size_t)N_NODES * ATOM * 4);
  _Float16* h16     = (_Float16*)alloc((size_t)N_NODES * ATOM * 2);
  _Float16* m16     = (_Float16*)alloc((size_t)N_NODES * ATOM * 2);
  float*    gi      = (float*)   alloc((size_t)N_NODES * 192 * 4);
  float*    gh      = (float*)   alloc((size_t)N_NODES * 192 * 4);
  float*    agg     = (float*)   alloc((size_t)N_NODES * ATOM * 4);
  float*    deg     = (float*)   alloc((size_t)N_NODES * 4);
  _Float16* Hedge   = (_Float16*)alloc((size_t)N_EDGES * CONV_DIM * 2);
  _Float16* Wedges  = (_Float16*)alloc((size_t)N_EDGES * ATOM * ATOM * 2); // 328 MB
  _Float16* WlinT   = (_Float16*)alloc(64 * 64 * 2);
  _Float16* We2T    = (_Float16*)alloc((size_t)4096 * 128 * 2);
  _Float16* WihF    = (_Float16*)alloc(192 * 64 * 2);
  _Float16* WhhF    = (_Float16*)alloc(192 * 64 * 2);
  _Float16* WcatF   = (_Float16*)alloc(256 * 192 * 2);
  float*    bcat    = (float*)   alloc(256 * 4);
  _Float16* Wm1T    = (_Float16*)alloc((size_t)MLP1 * 128 * 2);
  _Float16* Wm2T    = (_Float16*)alloc((size_t)MLP2 * MLP1 * 2);
  float*    qstar   = (float*)   alloc((size_t)N_GRAPHS * 128 * 4);
  float*    hl      = (float*)   alloc((size_t)N_GRAPHS * 64 * 4);
  float*    cl      = (float*)   alloc((size_t)N_GRAPHS * 64 * 4);
  _Float16* xcat16  = (_Float16*)alloc((size_t)N_GRAPHS * 192 * 2);
  float*    gbuf    = (float*)   alloc((size_t)N_GRAPHS * 256 * 4);
  float*    ebuf    = (float*)   alloc((size_t)N_NODES * 4);
  unsigned* emax    = (unsigned*)alloc((size_t)N_GRAPHS * 4);
  float*    zden    = (float*)   alloc((size_t)N_GRAPHS * 4);
  _Float16* y16     = (_Float16*)alloc((size_t)N_GRAPHS * 128 * 2);
  _Float16* y1_16   = (_Float16*)alloc((size_t)N_GRAPHS * MLP1 * 2);
  float*    y2      = (float*)   alloc((size_t)N_GRAPHS * MLP2 * 4);

  const int B = 256;
  auto gemm_blocks = [&](int M, int N) { return cdiv((long long)(M / 16) * (N / 16) * 32, B); };

  // --- weight prep (f16 / transposed for WMMA B operand [N,K]) ---
  k_transpose_f16<<<cdiv(64 * 64, B), B, 0, stream>>>(W_lin, WlinT, 64, 64);
  k_transpose_f16<<<cdiv(128 * 4096, B), B, 0, stream>>>(W_e2, We2T, 128, 4096);
  k_f32_to_f16<<<cdiv(192 * 64, B), B, 0, stream>>>(gWih, WihF, 192 * 64);
  k_f32_to_f16<<<cdiv(192 * 64, B), B, 0, stream>>>(gWhh, WhhF, 192 * 64);
  k_wcat<<<cdiv(256 * 192, B), B, 0, stream>>>(lWih, lWhh, lbih, lbhh, WcatF, bcat);
  k_transpose_f16<<<cdiv(128 * MLP1, B), B, 0, stream>>>(W_m1, Wm1T, 128, MLP1);
  k_transpose_f16<<<cdiv(MLP1 * MLP2, B), B, 0, stream>>>(W_m2, Wm2T, MLP1, MLP2);

  // --- node linear: h = relu(x_feat @ W_lin + b_lin) ---
  k_f32_to_f16<<<cdiv(N_NODES * ATOM, B), B, 0, stream>>>(x_feat, xfeat16, N_NODES * ATOM);
  k_wmma_gemm<true, true, true><<<gemm_blocks(N_NODES, ATOM), B, 0, stream>>>(
      xfeat16, WlinT, b_lin, h32, h16, N_NODES, ATOM, ATOM);

  // --- edge MLP: W_edges = (relu(ea@We1+b) @ We2 + b2), stored f16 ---
  k_edge1<<<cdiv(N_EDGES * CONV_DIM, B), B, 0, stream>>>(edge_attr, W_e1, b_e1, Hedge);
  k_wmma_gemm<false, false, true><<<gemm_blocks(N_EDGES, ATOM * ATOM), B, 0, stream>>>(
      Hedge, We2T, b_e2, nullptr, Wedges, N_EDGES, ATOM * ATOM, CONV_DIM);

  // --- degrees ---
  k_zero<<<cdiv(N_NODES, B), B, 0, stream>>>(deg, N_NODES);
  k_deg<<<cdiv(N_EDGES, B), B, 0, stream>>>(dstI, deg, N_EDGES);

  // --- 3 message-passing + GRU rounds ---
  for (int r = 0; r < 3; ++r) {
    k_zero<<<cdiv(N_NODES * ATOM, B), B, 0, stream>>>(agg, N_NODES * ATOM);
    k_message<<<cdiv((long long)N_EDGES * 32, B), B, 0, stream>>>(h32, Wedges, srcI, dstI, agg, N_EDGES);
    k_mprep<<<cdiv(N_NODES * ATOM, B), B, 0, stream>>>(agg, deg, b_conv, m16);
    k_wmma_gemm<false, true, false><<<gemm_blocks(N_NODES, 192), B, 0, stream>>>(
        m16, WihF, gbih, gi, nullptr, N_NODES, 192, ATOM);
    k_wmma_gemm<false, true, false><<<gemm_blocks(N_NODES, 192), B, 0, stream>>>(
        h16, WhhF, gbhh, gh, nullptr, N_NODES, 192, ATOM);
    k_gru<<<cdiv(N_NODES * ATOM, B), B, 0, stream>>>(gi, gh, h32, h16);
  }

  // --- Set2Set pooling ---
  k_zero<<<cdiv(N_GRAPHS * 128, B), B, 0, stream>>>(qstar, N_GRAPHS * 128);
  k_zero<<<cdiv(N_GRAPHS * 64, B), B, 0, stream>>>(hl, N_GRAPHS * 64);
  k_zero<<<cdiv(N_GRAPHS * 64, B), B, 0, stream>>>(cl, N_GRAPHS * 64);
  for (int s = 0; s < 3; ++s) {
    k_xcat<<<cdiv(N_GRAPHS * 192, B), B, 0, stream>>>(qstar, hl, xcat16);
    k_wmma_gemm<false, true, false><<<gemm_blocks(N_GRAPHS, 256), B, 0, stream>>>(
        xcat16, WcatF, bcat, gbuf, nullptr, N_GRAPHS, 256, 192);
    k_lstm<<<cdiv(N_GRAPHS * 64, B), B, 0, stream>>>(gbuf, cl, hl, qstar);
    k_zero<<<cdiv(N_GRAPHS, B), B, 0, stream>>>((float*)emax, N_GRAPHS);
    k_zero<<<cdiv(N_GRAPHS, B), B, 0, stream>>>(zden, N_GRAPHS);
    k_dot<<<cdiv((long long)N_NODES * 32, B), B, 0, stream>>>(h32, hl, batch, ebuf, emax);
    k_expsum<<<cdiv(N_NODES, B), B, 0, stream>>>(ebuf, emax, batch, zden);
    k_attn<<<cdiv((long long)N_NODES * 32, B), B, 0, stream>>>(ebuf, zden, batch, h32, qstar);
  }

  // --- BN + MLP head ---
  k_bn<<<1, 128, 0, stream>>>(qstar, bn_g, bn_b, y16);
  k_wmma_gemm<true, false, true><<<gemm_blocks(N_GRAPHS, MLP1), B, 0, stream>>>(
      y16, Wm1T, b_m1, nullptr, y1_16, N_GRAPHS, MLP1, 128);
  k_wmma_gemm<true, true, false><<<gemm_blocks(N_GRAPHS, MLP2), B, 0, stream>>>(
      y1_16, Wm2T, b_m2, y2, nullptr, N_GRAPHS, MLP2, MLP1);
  k_head<<<cdiv(N_GRAPHS * OUTD, B), B, 0, stream>>>(y2, W_p, b_p, out);
}
